// CasualSelfAttention_1425929142361
// MI455X (gfx1250) — matrix operand order repaired and checked by reading, not compiled, not run on hardware
//
#include <hip/hip_runtime.h>

// ---------------------------------------------------------------------------
// Causal self-attention for MI455X (gfx1250, wave32, WMMA + TDM).
// cvt(x) | transpose-cvt(W) -> QKV gemm (bf16 WMMA, f32 acc, double-buffered
// fragments) -> flash attention (TDM tensor_load_to_lds double-buffered K/V
// tiles, streaming softmax) -> projection gemm -> f32 out.
// ---------------------------------------------------------------------------

typedef __attribute__((ext_vector_type(16))) __bf16 v16bf;
typedef __attribute__((ext_vector_type(8)))  __bf16 v8bf;
typedef __attribute__((ext_vector_type(8)))  float  v8f;
typedef __attribute__((ext_vector_type(4)))  unsigned int u32x4;
typedef __attribute__((ext_vector_type(8)))  int i32x8;
typedef __attribute__((ext_vector_type(4)))  int i32x4;

#define B_  2
#define T_  2048
#define C_  1024
#define H_  16
#define HD_ 64
#define M_  (B_ * T_)        // 4096 rows of x
#define N1_ (3 * C_)         // 3072 qkv cols
#define SCALE_ 0.0125f       // 0.1 / sqrt(64)

__device__ __forceinline__ __bf16 f2bf(float f) {
  unsigned u = __float_as_uint(f);
  u = (u + 0x7FFFu + ((u >> 16) & 1u)) >> 16;   // round-to-nearest-even
  unsigned short s = (unsigned short)u;
  __bf16 r;
  __builtin_memcpy(&r, &s, 2);
  return r;
}

__device__ __forceinline__ v16bf pack16(v8bf a, v8bf b) {
  v16bf r;
#pragma unroll
  for (int i = 0; i < 8; ++i) { r[i] = a[i]; r[i + 8] = b[i]; }
  return r;
}

// ---------------------------------------------------------------------------
// TDM: issue a 2D bf16 tile load (tile_d1 rows x tile_d0 elems, row stride
// stride0 elems) from global into LDS at byte offset lds_off.
// D# layout per CDNA5 ISA; 6-arg builtin (clang-23
// amdgpu-toolchain lane: g0, g1, g2, g3, g4, cpol).
// ---------------------------------------------------------------------------
__device__ __forceinline__ void tdm_load_2d(unsigned lds_off, const void* gptr,
                                            unsigned tile_d0, unsigned tile_d1,
                                            unsigned tensor_d0, unsigned tensor_d1,
                                            unsigned stride0) {
  unsigned long long ga = (unsigned long long)gptr;
  u32x4 g0;
  g0[0] = 1u;                                        // count=1, load, user mode
  g0[1] = lds_off;                                   // lds_addr (bytes)
  g0[2] = (unsigned)ga;                              // global_addr[31:0]
  g0[3] = (unsigned)((ga >> 32) & 0x01FFFFFFu) | (2u << 30);  // addr[56:32]|type=2
  i32x8 g1;
  g1[0] = 1 << 16;                                   // data_size=1 (2 bytes)
  g1[1] = (int)((tensor_d0 & 0xFFFFu) << 16);        // tensor_dim0 lo16
  g1[2] = (int)((tensor_d0 >> 16) | ((tensor_d1 & 0xFFFFu) << 16));
  g1[3] = (int)((tensor_d1 >> 16) | (tile_d0 << 16));// tensor_dim1 hi | tile_dim0
  g1[4] = (int)(tile_d1 & 0xFFFFu);                  // tile_dim1 (tile_dim2=0)
  g1[5] = (int)stride0;                              // tensor_dim0_stride lo32
  g1[6] = 0;
  g1[7] = 0;
  i32x4 z4 = {};
  i32x8 z8 = {};
  __builtin_amdgcn_tensor_load_to_lds(g0, g1, z4, z4, z8, 0);
}

// ---------------------------------------------------------------------------
// Conversion / transpose kernels
// ---------------------------------------------------------------------------
__global__ void cvt_f32_bf16(const float* __restrict__ src,
                             __bf16* __restrict__ dst, int n) {
  int i = blockIdx.x * 256 + threadIdx.x;
  if (i < n) dst[i] = f2bf(src[i]);
}

// src[K][N] (row major f32) -> dst[N][K] (row major bf16)
__global__ void transpose_cvt(const float* __restrict__ src,
                              __bf16* __restrict__ dst, int K, int N) {
  int idx = blockIdx.x * 256 + threadIdx.x;
  if (idx < K * N) {
    int k = idx / N;
    int n = idx - k * N;
    dst[(size_t)n * K + k] = f2bf(src[idx]);
  }
}

// ---------------------------------------------------------------------------
// GEMM helpers: one wave = 16(M) x 64(N) tile, 4 f32 accumulators, register
// double-buffered fragments so vmem overlaps the matrix pipe.
// A: bf16 [M][K] row major; Bt: bf16 [N][K] row major (B transposed) so every
// B fragment is two runs of 8 contiguous elems, 16 apart in k (2 x b128).
// ---------------------------------------------------------------------------
__device__ __forceinline__ void load_frags(const __bf16* arow, const __bf16* bcol0,
                                           int ldb, int k0, int hi,
                                           v16bf& a, v16bf b[4]) {
  const __bf16* ap = arow + k0 + hi * 8;
  a = pack16(*(const v8bf*)ap, *(const v8bf*)(ap + 16));
#pragma unroll
  for (int nc = 0; nc < 4; ++nc) {
    const __bf16* bp = bcol0 + (size_t)nc * 16 * ldb + k0 + hi * 8;
    b[nc] = pack16(*(const v8bf*)bp, *(const v8bf*)(bp + 16));
  }
}

__device__ __forceinline__ void wmma4(v8f acc[4], v16bf a, const v16bf b[4]) {
#pragma unroll
  for (int nc = 0; nc < 4; ++nc)
    acc[nc] = __builtin_amdgcn_wmma_f32_16x16x32_bf16(
        false, a, false, b[nc], (short)0, acc[nc], false, false);
}

__device__ __forceinline__ void gemm_tile_16x64(
    const __bf16* __restrict__ Atile, int lda,
    const __bf16* __restrict__ Bttile, int ldb,
    int K, int lane, v8f acc[4]) {
  const int lo = lane & 15, hi = lane >> 4;
  const __bf16* arow  = Atile  + (size_t)lo * lda;
  const __bf16* bcol0 = Bttile + (size_t)lo * ldb;

  v16bf a0, a1, b0[4], b1[4];
  load_frags(arow, bcol0, ldb, 0, hi, a0, b0);
  for (int k0 = 0; k0 < K; k0 += 64) {
    __builtin_prefetch(arow + k0 + 512, 0, 0);          // global_prefetch_b8
    load_frags(arow, bcol0, ldb, k0 + 32, hi, a1, b1);  // next half-step
    wmma4(acc, a0, b0);
    if (k0 + 64 < K) load_frags(arow, bcol0, ldb, k0 + 64, hi, a0, b0);
    wmma4(acc, a1, b1);
  }
}

// ---------------------------------------------------------------------------
// QKV projection: qkv = x @ W_attn + b_attn, scattered into
//   Q,K : bf16 [B*H][T][64]   (row major per head)
//   Vt  : bf16 [B*H][64][T]   (transposed per head, for PV B-fragments / TDM)
// ---------------------------------------------------------------------------
__global__ __launch_bounds__(128) void qkv_gemm(
    const __bf16* __restrict__ xbf, const __bf16* __restrict__ WattnT,
    const float* __restrict__ b_attn,
    __bf16* __restrict__ Q, __bf16* __restrict__ Kh, __bf16* __restrict__ Vt) {
  const int lane = threadIdx.x, lo = lane & 15, hi = lane >> 4;
  const int mtile = blockIdx.x;                       // 0..255
  const int ntile = blockIdx.y * 4 + threadIdx.y;     // 0..47 (64-wide)

  v8f acc[4] = {};
  gemm_tile_16x64(xbf + (size_t)mtile * 16 * C_, C_,
                  WattnT + (size_t)ntile * 64 * C_, C_, C_, lane, acc);

#pragma unroll
  for (int nc = 0; nc < 4; ++nc) {
    int n = ntile * 64 + nc * 16 + lo;
    float bias = b_attn[n];
    int sec = n >> 10;          // 0:q 1:k 2:v
    int c = n & (C_ - 1);
    int h = c >> 6;
    int d = c & 63;
#pragma unroll
    for (int r = 0; r < 8; ++r) {
      int m = mtile * 16 + r + 8 * hi;
      int b = m >> 11;
      int t = m & (T_ - 1);
      __bf16 v = f2bf(acc[nc][r] + bias);
      size_t bh = (size_t)(b * H_ + h);
      if (sec == 0)      Q [(bh * T_ + t) * HD_ + d] = v;
      else if (sec == 1) Kh[(bh * T_ + t) * HD_ + d] = v;
      else               Vt[(bh * HD_ + d) * T_ + t] = v;
    }
  }
}

// ---------------------------------------------------------------------------
// Flash attention: 1 wave per (b,h, 16-query tile). TDM double-buffers K/V
// 32-key tiles in LDS; per chunk: 4 QK^T WMMAs -> S in LDS -> online softmax
// -> P (bf16) in LDS -> 4 PV WMMAs into 16x64 f32 accumulators.
// ---------------------------------------------------------------------------
__global__ __launch_bounds__(32) void attn_kernel(
    const __bf16* __restrict__ Q, const __bf16* __restrict__ Kh,
    const __bf16* __restrict__ Vt, __bf16* __restrict__ ybf) {
  __shared__ __align__(16) __bf16 Klds[2][32][64];   // [buf][key][d]
  __shared__ __align__(16) __bf16 Vlds[2][64][32];   // [buf][d][key]
  __shared__ __align__(64) float  Slds[16][32];
  __shared__ __align__(64) __bf16 Plds[16][32];
  __shared__ float alpha_lds[16];
  __shared__ float linv_lds[16];

  const int lane = threadIdx.x, lo = lane & 15, hi = lane >> 4;
  const int qt = blockIdx.x;           // 0..127
  const int bh = blockIdx.y;           // 0..31 = b*H + h

  const __bf16* Qp = Q  + (size_t)bh * T_ * HD_;
  const __bf16* Kp = Kh + (size_t)bh * T_ * HD_;
  const __bf16* Vp = Vt + (size_t)bh * HD_ * T_;

  const unsigned klds_off = (unsigned)(unsigned long long)(const void*)&Klds[0][0][0];
  const unsigned vlds_off = (unsigned)(unsigned long long)(const void*)&Vlds[0][0][0];
  const unsigned KBUF = 32u * 64u * 2u;   // 4KB per buffer

  // Q A-fragments, d in [0,32) and [32,64)
  const __bf16* qrowp = Qp + (size_t)(qt * 16 + lo) * HD_;
  v16bf aq[2];
#pragma unroll
  for (int j = 0; j < 2; ++j) {
    const __bf16* ap = qrowp + j * 32 + hi * 8;
    aq[j] = pack16(*(const v8bf*)ap, *(const v8bf*)(ap + 16));
  }

  float m_run = -3.0e38f, l_run = 0.0f;
  v8f acc[4] = {};
  const int qmax = qt * 16 + 15;

  // prologue: TDM-load chunk 0 into buffer 0
  tdm_load_2d(klds_off, Kp, 64, 32, 64, (unsigned)T_, 64);
  tdm_load_2d(vlds_off, Vp, 32, 64, (unsigned)T_, 64, (unsigned)T_);

  int buf = 0;
  for (int kc = 0; kc <= qmax; kc += 32) {
    const bool has_next = (kc + 32 <= qmax);
    if (has_next) {   // TDM-prefetch next chunk into the other buffer
      int nb = buf ^ 1;
      tdm_load_2d(klds_off + nb * KBUF, Kp + (size_t)(kc + 32) * HD_,
                  64, 32, 64, (unsigned)(T_ - kc - 32), 64);
      tdm_load_2d(vlds_off + nb * KBUF, Vp + kc + 32,
                  32, 64, (unsigned)(T_ - kc - 32), 64, (unsigned)T_);
      __builtin_amdgcn_s_wait_tensorcnt(2);   // current chunk's 2 xfers done
    } else {
      __builtin_amdgcn_s_wait_tensorcnt(0);
    }
    asm volatile("" ::: "memory");

    // ---- S = Q K^T for keys [kc, kc+32), two 16-key subtiles ----
    v8f s[2] = {};
#pragma unroll
    for (int sub = 0; sub < 2; ++sub) {
      const __bf16* kp = &Klds[buf][sub * 16 + lo][0];
#pragma unroll
      for (int j = 0; j < 2; ++j) {
        const __bf16* bp = kp + j * 32 + hi * 8;
        v16bf bk = pack16(*(const v8bf*)bp, *(const v8bf*)(bp + 16));
        s[sub] = __builtin_amdgcn_wmma_f32_16x16x32_bf16(
            false, aq[j], false, bk, (short)0, s[sub], false, false);
      }
    }
    // C layout: lane=key col (lo), reg r = q row r+8*hi
#pragma unroll
    for (int r = 0; r < 8; ++r) {
      Slds[r + 8 * hi][lo]      = s[0][r] * SCALE_;
      Slds[r + 8 * hi][16 + lo] = s[1][r] * SCALE_;
    }
    __syncthreads();

    // ---- online softmax, one q row per lane (lanes 0..15) ----
    if (lane < 16) {
      int qg = qt * 16 + lane;
      int kvalid = qg - kc + 1;            // causal: keys kc..qg
      if (kvalid > 32) kvalid = 32;        // >=1 always (kc <= qt*16 <= qg)
      float mx = m_run;
      for (int j = 0; j < kvalid; ++j) mx = fmaxf(mx, Slds[lane][j]);
      float alpha = __expf(m_run - mx);    // first chunk: underflows to 0
      float sum = 0.0f;
      for (int j = 0; j < 32; ++j) {
        float p = (j < kvalid) ? __expf(Slds[lane][j] - mx) : 0.0f;
        Plds[lane][j] = f2bf(p);
        sum += p;
      }
      l_run = l_run * alpha + sum;
      m_run = mx;
      alpha_lds[lane] = alpha;
    }
    __syncthreads();

    // ---- rescale accumulators, then O += P @ V ----
    float al[8];
#pragma unroll
    for (int r = 0; r < 8; ++r) al[r] = alpha_lds[r + 8 * hi];
#pragma unroll
    for (int dc = 0; dc < 4; ++dc)
#pragma unroll
      for (int r = 0; r < 8; ++r) acc[dc][r] *= al[r];

    const __bf16* prow = &Plds[lo][0];
    v16bf ap = pack16(*(const v8bf*)(prow + hi * 8),
                      *(const v8bf*)(prow + 16 + hi * 8));
#pragma unroll
    for (int dc = 0; dc < 4; ++dc) {
      const __bf16* vp = &Vlds[buf][dc * 16 + lo][hi * 8];
      v16bf bv = pack16(*(const v8bf*)vp, *(const v8bf*)(vp + 16));
      acc[dc] = __builtin_amdgcn_wmma_f32_16x16x32_bf16(
          false, ap, false, bv, (short)0, acc[dc], false, false);
    }
    __syncthreads();
    buf ^= 1;
  }

  if (lane < 16) linv_lds[lane] = 1.0f / l_run;
  __syncthreads();

  const int b = bh >> 4, h = bh & 15;
#pragma unroll
  for (int r = 0; r < 8; ++r) {
    int m = b * T_ + qt * 16 + r + 8 * hi;
    float li = linv_lds[r + 8 * hi];
#pragma unroll
    for (int dc = 0; dc < 4; ++dc)
      ybf[(size_t)m * C_ + h * HD_ + dc * 16 + lo] = f2bf(acc[dc][r] * li);
  }
}

// ---------------------------------------------------------------------------
// Output projection: out = y @ W_proj + b_proj (f32 out)
// ---------------------------------------------------------------------------
__global__ __launch_bounds__(128) void proj_gemm(
    const __bf16* __restrict__ ybf, const __bf16* __restrict__ WprojT,
    const float* __restrict__ b_proj, float* __restrict__ out) {
  const int lane = threadIdx.x, lo = lane & 15, hi = lane >> 4;
  const int mtile = blockIdx.x;                    // 0..255
  const int ntile = blockIdx.y * 4 + threadIdx.y;  // 0..15

  v8f acc[4] = {};
  gemm_tile_16x64(ybf + (size_t)mtile * 16 * C_, C_,
                  WprojT + (size_t)ntile * 64 * C_, C_, C_, lane, acc);

#pragma unroll
  for (int nc = 0; nc < 4; ++nc) {
    int n = ntile * 64 + nc * 16 + lo;
    float bias = b_proj[n];
#pragma unroll
    for (int r = 0; r < 8; ++r) {
      int m = mtile * 16 + r + 8 * hi;
      out[(size_t)m * C_ + n] = acc[nc][r] + bias;
    }
  }
}

// ---------------------------------------------------------------------------
extern "C" void kernel_launch(void* const* d_in, const int* in_sizes, int n_in,
                              void* d_out, int out_size, void* d_ws, size_t ws_size,
                              hipStream_t stream) {
  const float* x      = (const float*)d_in[0];   // [B,T,C]
  const float* W_attn = (const float*)d_in[1];   // [C,3C]
  const float* b_attn = (const float*)d_in[2];   // [3C]
  const float* W_proj = (const float*)d_in[3];   // [C,C]
  const float* b_proj = (const float*)d_in[4];   // [C]
  float* out = (float*)d_out;

  char* w = (char*)d_ws;
  __bf16* xbf    = (__bf16*)(w);                    //  8.39 MB
  __bf16* WattnT = (__bf16*)(w + (8u  << 20));      //  6.29 MB  [3072][1024]
  __bf16* WprojT = (__bf16*)(w + (14u << 20));      //  2.10 MB  [1024][1024]
  __bf16* Qb     = (__bf16*)(w + (16u << 20));      //  8.39 MB  [BH][T][64]
  __bf16* Kb     = (__bf16*)(w + (24u << 20));      //  8.39 MB  [BH][T][64]
  __bf16* Vt     = (__bf16*)(w + (32u << 20));      //  8.39 MB  [BH][64][T]
  __bf16* ybf    = (__bf16*)(w + (40u << 20));      //  8.39 MB  [M][C]

  // 1) precision staging
  cvt_f32_bf16<<<(M_ * C_ + 255) / 256, 256, 0, stream>>>(x, xbf, M_ * C_);
  transpose_cvt<<<(C_ * N1_ + 255) / 256, 256, 0, stream>>>(W_attn, WattnT, C_, N1_);
  transpose_cvt<<<(C_ * C_ + 255) / 256, 256, 0, stream>>>(W_proj, WprojT, C_, C_);

  // 2) QKV projection
  qkv_gemm<<<dim3(M_ / 16, N1_ / 256), dim3(32, 4), 0, stream>>>(
      xbf, WattnT, b_attn, Qb, Kb, Vt);

  // 3) flash attention (TDM-pipelined)
  attn_kernel<<<dim3(T_ / 16, B_ * H_), dim3(32, 1), 0, stream>>>(Qb, Kb, Vt, ybf);

  // 4) output projection
  proj_gemm<<<dim3(M_ / 16, C_ / 256), dim3(32, 4), 0, stream>>>(
      ybf, WprojT, b_proj, out);
}
